// MultiHeadSelfAttention_19696720019827
// MI455X (gfx1250) — matrix operand order repaired and checked by reading, not compiled, not run on hardware
//
#include <hip/hip_runtime.h>
#include <stdint.h>

typedef __attribute__((ext_vector_type(16))) __bf16 v16bf;
typedef __attribute__((ext_vector_type(8)))  float  v8f;

#define NH   16
#define HD   64
#define SEQ  2048
#define HID  1024
#define BATCH 2

union Frag {
  uint4 q[2];               // two 16B halves
  v16bf v;                  // 16 bf16 = 8 VGPRs
  __bf16 b[16];
  unsigned short s[16];
};

// native f32 -> bf16 (v_cvt_*bf16_f32), raw bits for stores
__device__ __forceinline__ unsigned short f2bf(float f) {
  union { __bf16 b; unsigned short s; } x;
  x.b = (__bf16)f;
  return x.s;
}

__device__ __forceinline__ v8f wmma_bf16(const Frag& a, const Frag& b, v8f c) {
  return __builtin_amdgcn_wmma_f32_16x16x32_bf16(
      false, a.v, false, b.v, (short)0, c, false, false);
}

// Load one 16x32 A fragment from a bf16 row-major matrix (row stride HID).
__device__ __forceinline__ void loadA_bf16(Frag& a, const unsigned short* xrow,
                                           int k, int h2) {
  a.q[0] = *reinterpret_cast<const uint4*>(xrow + k + 16 * h2);
  a.q[1] = *reinterpret_cast<const uint4*>(xrow + k + 16 * h2 + 8);
}

// ---------------- bulk f32 -> bf16 convert (8 elements / thread) ----------------
__global__ void __launch_bounds__(256)
convert_bf16_kernel(const float* __restrict__ src, unsigned short* __restrict__ dst) {
  const size_t i = ((size_t)blockIdx.x * 256 + threadIdx.x) * 8;
  float4 f0 = *reinterpret_cast<const float4*>(src + i);
  float4 f1 = *reinterpret_cast<const float4*>(src + i + 4);
  Frag a;
  a.b[0] = (__bf16)f0.x; a.b[1] = (__bf16)f0.y; a.b[2] = (__bf16)f0.z; a.b[3] = (__bf16)f0.w;
  a.b[4] = (__bf16)f1.x; a.b[5] = (__bf16)f1.y; a.b[6] = (__bf16)f1.z; a.b[7] = (__bf16)f1.w;
  *reinterpret_cast<uint4*>(dst + i) = a.q[0];
}

// ---------------- weight transpose + f32->bf16 convert ----------------
// W: K x N (f32) -> WT: N x K (bf16)
__global__ void __launch_bounds__(256)
transpose_bf16_kernel(const float* __restrict__ W, unsigned short* __restrict__ WT,
                      int K, int N) {
  __shared__ unsigned short tile[32][33];
  const int tx = threadIdx.x;            // 0..31
  const int ty = threadIdx.y;            // 0..7
  const int k0 = blockIdx.y * 32;
  const int n0 = blockIdx.x * 32;
  for (int i = ty; i < 32; i += 8)
    tile[i][tx] = f2bf(W[(size_t)(k0 + i) * N + (n0 + tx)]);
  __syncthreads();
  for (int i = ty; i < 32; i += 8)
    WT[(size_t)(n0 + i) * K + (k0 + tx)] = tile[tx][i];
}

// ---------------- QKV projection GEMM + head scatter ----------------
// Xb: (4096 x 1024) bf16, WT: (3072 x 1024) bf16, bias: 3072 f32
// Wave tile 32x64: two A fragments share four B fragments per k-step.
// Writes Qb,Kb as (B,h,S,d) bf16 (Q pre-scaled by 8), Vt as (B,h,d,S) bf16.
__global__ void __launch_bounds__(256)
qkv_gemm_kernel(const unsigned short* __restrict__ Xb,
                const unsigned short* __restrict__ WT,
                const float* __restrict__ bias,
                unsigned short* __restrict__ Qb, unsigned short* __restrict__ Kb,
                unsigned short* __restrict__ Vt) {
  const int ln  = threadIdx.x & 15;
  const int h2  = threadIdx.x >> 4;
  const int row0 = (blockIdx.y * 8 + threadIdx.y) * 32;   // 32-row wave tile
  const int col0 = blockIdx.x * 64;

  v8f acc[2][4] = {{{}, {}, {}, {}}, {{}, {}, {}, {}}};
  const unsigned short* xrow0 = Xb + (size_t)(row0 + ln) * HID;
  const unsigned short* xrow1 = Xb + (size_t)(row0 + 16 + ln) * HID;

  for (int k = 0; k < HID; k += 32) {
    Frag a0, a1;
    loadA_bf16(a0, xrow0, k, h2);
    loadA_bf16(a1, xrow1, k, h2);
    for (int j = 0; j < 4; ++j) {
      Frag b;
      const uint4* w4 = reinterpret_cast<const uint4*>(
          WT + (size_t)(col0 + 16 * j + ln) * HID + k + 16 * h2);
      b.q[0] = w4[0]; b.q[1] = w4[1];
      acc[0][j] = wmma_bf16(a0, b, acc[0][j]);
      acc[1][j] = wmma_bf16(a1, b, acc[1][j]);
    }
  }

  for (int j = 0; j < 4; ++j) {
    const int c = col0 + 16 * j + ln;
    const float bv = bias[c];
    const int region = c >> 10;         // 0=Q 1=K 2=V
    const int cc = c & 1023;
    const int head = cc >> 6, dd = cc & 63;
    for (int t = 0; t < 2; ++t) {
      for (int r = 0; r < 8; ++r) {
        const int row = row0 + 16 * t + r + 8 * h2;
        const int bb = row >> 11, sl = row & 2047;
        const float val = acc[t][j][r] + bv;
        if (region == 0)
          Qb[(((size_t)(bb * NH + head)) * SEQ + sl) * HD + dd] = f2bf(val * 8.0f);
        else if (region == 1)
          Kb[(((size_t)(bb * NH + head)) * SEQ + sl) * HD + dd] = f2bf(val);
        else
          Vt[(((size_t)(bb * NH + head)) * HD + dd) * SEQ + sl] = f2bf(val);
      }
    }
  }
}

// ---------------- flash attention (one wave = one 16-row Q tile) ----------------
// Output stored directly as bf16 (B*S, 1024): it is only consumed as the
// A-operand of the output projection GEMM.
__global__ void __launch_bounds__(256)
attn_kernel(const unsigned short* __restrict__ Qb, const unsigned short* __restrict__ Kb,
            const unsigned short* __restrict__ Vt, const unsigned char* __restrict__ mask,
            unsigned short* __restrict__ outB) {
  __shared__ unsigned short lds_p[8][16 * 32];   // per-wave P staging (bf16)
  const int ln = threadIdx.x & 15;
  const int h2 = threadIdx.x >> 4;
  const int wv = threadIdx.y;
  const int head = blockIdx.y, b = blockIdx.z;
  const int q0 = (blockIdx.x * 8 + wv) * 16;
  const size_t bh = (size_t)(b * NH + head);
  const unsigned short* Qh = Qb + bh * SEQ * HD;
  const unsigned short* Kh = Kb + bh * SEQ * HD;
  const unsigned short* Vh = Vt + bh * HD * SEQ;
  const unsigned char*  mk = mask + (size_t)b * SEQ;
  const float NEGINF = -__builtin_inff();

  // Q A-fragments for the two k-steps (d in [0,32), [32,64))
  Frag aq[2];
  {
    const unsigned short* qrow = Qh + (size_t)(q0 + ln) * HD;
    for (int s = 0; s < 2; ++s) {
      aq[s].q[0] = *reinterpret_cast<const uint4*>(qrow + 32 * s + 16 * h2);
      aq[s].q[1] = *reinterpret_cast<const uint4*>(qrow + 32 * s + 16 * h2 + 8);
    }
  }

  float mrow[8], lrow[8];
  v8f O[4] = {{}, {}, {}, {}};
  for (int r = 0; r < 8; ++r) { mrow[r] = NEGINF; lrow[r] = 0.0f; }

  for (int kc = 0; kc < SEQ; kc += 32) {
    // ---- scores: two 16x16 tiles over 32 keys ----
    v8f st[2];
    for (int t = 0; t < 2; ++t) {
      const unsigned short* krow = Kh + (size_t)(kc + 16 * t + ln) * HD;
      Frag b0, b1;
      const uint4* p0 = reinterpret_cast<const uint4*>(krow + 16 * h2);
      b0.q[0] = p0[0]; b0.q[1] = p0[1];
      const uint4* p1 = reinterpret_cast<const uint4*>(krow + 32 + 16 * h2);
      b1.q[0] = p1[0]; b1.q[1] = p1[1];
      v8f c = {};
      c = wmma_bf16(aq[0], b0, c);
      c = wmma_bf16(aq[1], b1, c);
      st[t] = c;
    }

    // ---- mask (each lane owns one key column per tile) ----
    for (int t = 0; t < 2; ++t) {
      const bool msk = mk[kc + 16 * t + ln] != 0;
      for (int r = 0; r < 8; ++r)
        st[t][r] = msk ? NEGINF : st[t][r];
    }

    // ---- online softmax (half-wave reductions match C/D row layout) ----
    float scale[8], rsum[8];
    for (int r = 0; r < 8; ++r) {
      float cm = fmaxf(st[0][r], st[1][r]);
      for (int off = 8; off; off >>= 1) cm = fmaxf(cm, __shfl_xor(cm, off, 16));
      const float nm = fmaxf(mrow[r], cm);
      scale[r] = (mrow[r] == NEGINF) ? 0.0f : __expf(mrow[r] - nm);
      mrow[r] = nm;
    }
    for (int r = 0; r < 8; ++r) {
      const float nm = mrow[r];
      const float p0 = (st[0][r] == NEGINF) ? 0.0f : __expf(st[0][r] - nm);
      const float p1 = (st[1][r] == NEGINF) ? 0.0f : __expf(st[1][r] - nm);
      const int m = r + 8 * h2;
      lds_p[wv][m * 32 + ln]      = f2bf(p0);
      lds_p[wv][m * 32 + 16 + ln] = f2bf(p1);
      float rs = p0 + p1;
      for (int off = 8; off; off >>= 1) rs += __shfl_xor(rs, off, 16);
      rsum[r] = rs;
    }
    for (int r = 0; r < 8; ++r) {
      lrow[r] = lrow[r] * scale[r] + rsum[r];
      for (int j = 0; j < 4; ++j) O[j][r] *= scale[r];
    }

    // ---- reload P in A-layout from LDS ----
    Frag pa;
    {
      const unsigned short* pr = &lds_p[wv][ln * 32];
      pa.q[0] = *reinterpret_cast<const uint4*>(pr + 16 * h2);
      pa.q[1] = *reinterpret_cast<const uint4*>(pr + 16 * h2 + 8);
    }

    // ---- P @ V : four 16-wide d blocks, gemm-K = 32 keys ----
    for (int j = 0; j < 4; ++j) {
      Frag vb;
      const uint4* v4 = reinterpret_cast<const uint4*>(
          Vh + (size_t)(16 * j + ln) * SEQ + kc + 16 * h2);
      vb.q[0] = v4[0]; vb.q[1] = v4[1];
      O[j] = wmma_bf16(pa, vb, O[j]);
    }
  }

  // ---- normalize and store bf16 (B*S, 1024) ----
  for (int r = 0; r < 8; ++r) {
    const float inv = 1.0f / lrow[r];
    const int row = q0 + r + 8 * h2;
    unsigned short* orow = outB + (size_t)(b * SEQ + row) * HID + head * HD;
    for (int j = 0; j < 4; ++j)
      orow[16 * j + ln] = f2bf(O[j][r] * inv);
  }
}

// ---------------- output projection GEMM (32x64 wave tile) ----------------
__global__ void __launch_bounds__(256)
out_gemm_kernel(const unsigned short* __restrict__ Xb,
                const unsigned short* __restrict__ WoT,
                const float* __restrict__ bo, float* __restrict__ Y) {
  const int ln = threadIdx.x & 15;
  const int h2 = threadIdx.x >> 4;
  const int row0 = (blockIdx.y * 8 + threadIdx.y) * 32;
  const int col0 = blockIdx.x * 64;
  v8f acc[2][4] = {{{}, {}, {}, {}}, {{}, {}, {}, {}}};
  const unsigned short* xrow0 = Xb + (size_t)(row0 + ln) * HID;
  const unsigned short* xrow1 = Xb + (size_t)(row0 + 16 + ln) * HID;

  for (int k = 0; k < HID; k += 32) {
    Frag a0, a1;
    loadA_bf16(a0, xrow0, k, h2);
    loadA_bf16(a1, xrow1, k, h2);
    for (int j = 0; j < 4; ++j) {
      Frag b;
      const uint4* w4 = reinterpret_cast<const uint4*>(
          WoT + (size_t)(col0 + 16 * j + ln) * HID + k + 16 * h2);
      b.q[0] = w4[0]; b.q[1] = w4[1];
      acc[0][j] = wmma_bf16(a0, b, acc[0][j]);
      acc[1][j] = wmma_bf16(a1, b, acc[1][j]);
    }
  }

  for (int j = 0; j < 4; ++j) {
    const int c = col0 + 16 * j + ln;
    const float bv = bo[c];
    for (int t = 0; t < 2; ++t) {
      for (int r = 0; r < 8; ++r) {
        const int row = row0 + 16 * t + r + 8 * h2;
        Y[(size_t)row * HID + c] = acc[t][j][r] + bv;
      }
    }
  }
}

extern "C" void kernel_launch(void* const* d_in, const int* in_sizes, int n_in,
                              void* d_out, int out_size, void* d_ws, size_t ws_size,
                              hipStream_t stream) {
  const float*         qkv  = (const float*)d_in[0];
  const unsigned char* mask = (const unsigned char*)d_in[1];   // jax bool -> 1 byte
  const float*         Wqkv = (const float*)d_in[2];
  const float*         bqkv = (const float*)d_in[3];
  const float*         Wo   = (const float*)d_in[4];
  const float*         bo   = (const float*)d_in[5];
  float*               out  = (float*)d_out;

  char* ws = (char*)d_ws;
  unsigned short* WqkvT = (unsigned short*)(ws);                 //  6,291,456 B
  unsigned short* WoT   = (unsigned short*)(ws +  6291456);      //  2,097,152 B
  unsigned short* Qb    = (unsigned short*)(ws +  8388608);      //  8,388,608 B
  unsigned short* Kb    = (unsigned short*)(ws + 16777216);      //  8,388,608 B
  unsigned short* Vt    = (unsigned short*)(ws + 25165824);      //  8,388,608 B
  unsigned short* Xb    = (unsigned short*)(ws + 33554432);      //  8,388,608 B
  unsigned short* attnB = (unsigned short*)(ws + 41943040);      //  8,388,608 B
  (void)in_sizes; (void)n_in; (void)out_size; (void)ws_size;

  // 1. convert activations; transpose + convert weights to bf16
  convert_bf16_kernel<<<dim3(BATCH * SEQ * HID / 8 / 256), dim3(256), 0, stream>>>(
      qkv, Xb);
  transpose_bf16_kernel<<<dim3(3 * HID / 32, HID / 32), dim3(32, 8), 0, stream>>>(
      Wqkv, WqkvT, HID, 3 * HID);
  transpose_bf16_kernel<<<dim3(HID / 32, HID / 32), dim3(32, 8), 0, stream>>>(
      Wo, WoT, HID, HID);

  // 2. QKV projection (+bias, q*sqrt(d), head scatter, V transposed)
  qkv_gemm_kernel<<<dim3(3 * HID / 64, BATCH * SEQ / 256), dim3(32, 8), 0, stream>>>(
      Xb, WqkvT, bqkv, Qb, Kb, Vt);

  // 3. flash attention (bf16 output)
  attn_kernel<<<dim3(SEQ / 16 / 8, NH, BATCH), dim3(32, 8), 0, stream>>>(
      Qb, Kb, Vt, mask, attnB);

  // 4. output projection (+bias)
  out_gemm_kernel<<<dim3(HID / 64, BATCH * SEQ / 256), dim3(32, 8), 0, stream>>>(
      attnB, WoT, bo, out);
}
